// LGMTransformation_88751204204610
// MI455X (gfx1250) — matrix operand order repaired and checked by reading, not compiled, not run on hardware
//
#include <hip/hip_runtime.h>

// Problem constants
#define B_    16
#define CIN   64
#define COUT  64
#define H_    256
#define W_    256
#define MD    16            // modes per axis
#define KYM   32            // ky modes kept: 0..15 and 240..255
#define NMODE (KYM*MD)      // 512
#define KCONV (CIN*9)       // 576
#define KSTEPS (KCONV/32)   // 18

typedef __attribute__((ext_vector_type(16))) __bf16          v16bf;
typedef __attribute__((ext_vector_type(16))) unsigned short  v16u;
typedef __attribute__((ext_vector_type(8)))  unsigned short  v8u;
typedef __attribute__((ext_vector_type(8)))  float           v8f;

union Frag { v16u u; v16bf b; };

__device__ __forceinline__ unsigned short f2bf(float f) {
  union { float f; unsigned u; } v; v.f = f;
  unsigned r = v.u + 0x7FFFu + ((v.u >> 16) & 1u);   // round-to-nearest-even
  return (unsigned short)(r >> 16);
}

// A-fragment K index for 16-bit 16x32 A matrix (contiguous order: k = 16*(lane/16) + idx)
__device__ __forceinline__ int a_klocal(int lane, int idx) {
  return ((lane >> 4) * 16) + idx;
}

// ---------------------------------------------------------------------------
// K0a: pack conv weights into per-lane WMMA A-fragment order (bf16).
// layout: [otile(4)][kstep(18)][lane(32)][idx(16)]  -> 32B contiguous per lane
// ---------------------------------------------------------------------------
__global__ void pack_conv_w(const float* __restrict__ cw, unsigned short* __restrict__ wfrag) {
  int t = blockIdx.x * 256 + threadIdx.x;
  if (t >= 4 * KSTEPS * 32 * 16) return;
  int idx   = t & 15;
  int lane  = (t >> 4) & 31;
  int rest  = t >> 9;
  int kstep = rest % KSTEPS;
  int otile = rest / KSTEPS;
  int k = kstep * 32 + a_klocal(lane, idx);
  int o = otile * 16 + (lane & 15);
  int i = k / 9, r = k % 9, dy = r / 3, dx = r % 3;
  wfrag[t] = f2bf(cw[((o * CIN + i) * 3 + dy) * 3 + dx]);
}

// ---------------------------------------------------------------------------
// K0b: pack spectral weights to bf16, layout [mode][ri][o][i] (transposed so
// B-fragment runs of consecutive k=i are contiguous in LDS).
// mode = kym*16+kx ; kym<16 -> w1[ky=kym], kym>=16 -> w2[ky=kym-16]
// ---------------------------------------------------------------------------
__global__ void pack_spec_w(const float* __restrict__ w1r, const float* __restrict__ w1i,
                            const float* __restrict__ w2r, const float* __restrict__ w2i,
                            unsigned short* __restrict__ wspk) {
  long t = (long)blockIdx.x * 256 + threadIdx.x;
  if (t >= (long)NMODE * 2 * CIN * COUT) return;
  int i    = t & 63;
  int o    = (t >> 6) & 63;
  int ri   = (t >> 12) & 1;
  int mode = (int)(t >> 13);
  int kx = mode & 15, kym = mode >> 4;
  const float* src = (kym < 16) ? (ri ? w1i : w1r) : (ri ? w2i : w2r);
  int ky = (kym < 16) ? kym : (kym - 16);
  wspk[t] = f2bf(src[((i * COUT + o) * MD + ky) * MD + kx]);
}

// ---------------------------------------------------------------------------
// K1: fused forward partial DFT per (b,i): c[256,256] -> X[32ky,16kx] complex.
// Phase 1: x-DFT into LDS; Phase 2: y-DFT; store bf16 to xpk[mode][ri][b][i].
// ---------------------------------------------------------------------------
__global__ __launch_bounds__(256) void dft_fwd(const float* __restrict__ c,
                                               unsigned short* __restrict__ xpk) {
  int i = blockIdx.x, b = blockIdx.y;
  __shared__ float cosT[256], sinT[256];
  __shared__ float rows[16 * 257];        // padded pitch: no bank conflicts
  __shared__ float tmpx[256 * 33];        // [y][2*kx] pitch 33
  int t = threadIdx.x;
  { float s, co; __sincosf(6.283185307179586f * t / 256.f, &s, &co); cosT[t] = co; sinT[t] = s; }
  __syncthreads();
  const float* base = c + ((long)(b * CIN + i)) * H_ * W_;
  for (int g = 0; g < 16; ++g) {
    for (int f = t; f < 16 * 256; f += 256) {
      int yl = f >> 8, x = f & 255;
      rows[yl * 257 + x] = base[(g * 16 + yl) * 256 + x];
    }
    __syncthreads();
    int yl = t >> 4, kx = t & 15;
    float cr = 0.f, ci = 0.f;
    const float* rp = &rows[yl * 257];
    #pragma unroll 4
    for (int x = 0; x < 256; ++x) {
      float cv = rp[x]; int a = (kx * x) & 255;
      cr += cv * cosT[a];            // e^{-i theta}: cos - i sin
      ci -= cv * sinT[a];
    }
    tmpx[(g * 16 + yl) * 33 + kx * 2]     = cr;
    tmpx[(g * 16 + yl) * 33 + kx * 2 + 1] = ci;
    __syncthreads();
  }
  for (int m = t; m < NMODE; m += 256) {
    int kx = m & 15, kym = m >> 4;
    int ky = (kym < 16) ? kym : (224 + kym);     // 240..255 = negative freqs
    float xr = 0.f, xi = 0.f;
    #pragma unroll 4
    for (int y = 0; y < 256; ++y) {
      float tr = tmpx[y * 33 + kx * 2], ti = tmpx[y * 33 + kx * 2 + 1];
      int a = (ky * y) & 255;
      float co = cosT[a], si = sinT[a];
      xr += tr * co + ti * si;                   // (tr+i ti)*(cos - i sin)
      xi += ti * co - tr * si;
    }
    xpk[((long)(m * 2 + 0) * B_ + b) * CIN + i] = f2bf(xr);
    xpk[((long)(m * 2 + 1) * B_ + b) * CIN + i] = f2bf(xi);
  }
}

// ---------------------------------------------------------------------------
// K2: spectral mix, one block per mode. G[b,o] = sum_i X[b,i]*W[i,o] (complex)
// WMMA: M=16(batch) N=64(out, 4 wave-tiles) K=64(in, 2 k-steps of 32), bf16.
// Fragments assembled from contiguous LDS runs via b128 loads.
// ---------------------------------------------------------------------------
__global__ __launch_bounds__(128) void spectral(const unsigned short* __restrict__ xpk,
                                                const unsigned short* __restrict__ wspk,
                                                float* __restrict__ gout) {
  int mode = blockIdx.x;
  int kym = mode >> 4, kx = mode & 15;
  __shared__ __align__(32) unsigned short lx[2 * B_ * CIN];    // [ri][b][i]
  __shared__ __align__(32) unsigned short lw[2 * COUT * CIN];  // [ri][o][i]
  int t = threadIdx.x;
  for (int f = t; f < 2 * B_ * CIN; f += 128)   lx[f] = xpk[(long)mode * 2 * B_ * CIN + f];
  for (int f = t; f < 2 * COUT * CIN; f += 128) lw[f] = wspk[(long)mode * 2 * COUT * CIN + f];
  __syncthreads();
  int lane = t & 31, ntile = t >> 5;
  int m  = lane & 15;                    // batch row
  int o  = ntile * 16 + (lane & 15);     // out-channel col
  int lh = lane >> 4;
  v8f accR = {}, accI = {};
  #pragma unroll
  for (int ks = 0; ks < 2; ++ks) {
    // A fragments: two contiguous 8-element (16B) runs per lane
    int abase = m * 64 + ks * 32 + lh * 16;
    v8u arlo = *(const v8u*)(&lx[0 * 1024 + abase]);
    v8u arhi = *(const v8u*)(&lx[0 * 1024 + abase + 8]);
    v8u ailo = *(const v8u*)(&lx[1 * 1024 + abase]);
    v8u aihi = *(const v8u*)(&lx[1 * 1024 + abase + 8]);
    Frag ar, ai, aiN;
    ar.u = __builtin_shufflevector(arlo, arhi, 0,1,2,3,4,5,6,7,8,9,10,11,12,13,14,15);
    ai.u = __builtin_shufflevector(ailo, aihi, 0,1,2,3,4,5,6,7,8,9,10,11,12,13,14,15);
    aiN.u = ai.u ^ (v16u)(0x8000);                 // -Xi via bf16 sign flip
    // B fragments: 16 consecutive k=i at fixed o -> one 32B run
    int bbase = o * 64 + ks * 32 + lh * 16;
    Frag br, bi;
    br.u = *(const v16u*)(&lw[0 * 4096 + bbase]);
    bi.u = *(const v16u*)(&lw[1 * 4096 + bbase]);
    accR = __builtin_amdgcn_wmma_f32_16x16x32_bf16(false, ar.b,  false, br.b, (short)0, accR, false, false);
    accR = __builtin_amdgcn_wmma_f32_16x16x32_bf16(false, aiN.b, false, bi.b, (short)0, accR, false, false);
    accI = __builtin_amdgcn_wmma_f32_16x16x32_bf16(false, ar.b,  false, bi.b, (short)0, accI, false, false);
    accI = __builtin_amdgcn_wmma_f32_16x16x32_bf16(false, ai.b,  false, br.b, (short)0, accI, false, false);
  }
  #pragma unroll
  for (int v = 0; v < 8; ++v) {
    int b = v + 8 * lh;
    long idx = ((((long)(b * COUT + o)) * KYM + kym) * MD + kx) * 2;
    gout[idx]     = accR[v];
    gout[idx + 1] = accI[v];
  }
}

// ---------------------------------------------------------------------------
// K3: inverse y-DFT per (b,o): G[32ky,16kx] -> h[b][y][o][kx] complex (f32).
// ---------------------------------------------------------------------------
__global__ __launch_bounds__(256) void idft_y(const float* __restrict__ gout,
                                              float* __restrict__ hbuf) {
  int o = blockIdx.x, b = blockIdx.y;
  __shared__ float cosT[256], sinT[256];
  __shared__ float lg[KYM * MD * 2];
  int t = threadIdx.x;
  { float s, co; __sincosf(6.283185307179586f * t / 256.f, &s, &co); cosT[t] = co; sinT[t] = s; }
  const float* gp = gout + ((long)(b * COUT + o)) * KYM * MD * 2;
  for (int f = t; f < KYM * MD * 2; f += 256) lg[f] = gp[f];
  __syncthreads();
  int y = t;
  float* hp = hbuf + (((long)(b * H_ + y)) * COUT + o) * MD * 2;
  for (int kx = 0; kx < MD; ++kx) {
    float hr = 0.f, hi = 0.f;
    #pragma unroll
    for (int kym = 0; kym < KYM; ++kym) {
      int ky = (kym < 16) ? kym : (224 + kym);
      int a = (ky * y) & 255;
      float co = cosT[a], si = sinT[a];
      float gr = lg[(kym * MD + kx) * 2], gi = lg[(kym * MD + kx) * 2 + 1];
      hr += gr * co - gi * si;                 // e^{+i theta}
      hi += gr * si + gi * co;
    }
    hp[kx * 2] = hr; hp[kx * 2 + 1] = hi;
  }
}

// ---------------------------------------------------------------------------
// K4: fused 3x3 circular conv (implicit GEMM, bf16 WMMA) + on-the-fly irfft-x
// + mixing. Block = (b, y, 16-wide x-tile); 4 waves = 4 out-channel tiles.
// GEMM: M=16(out ch tile) x N=16(pixels) x K=576, 18 k-steps.
// B-matrix pre-expanded in LDS as BT[pix][k] so each wave's B fragment is one
// aligned 32B contiguous load (2 x ds_load_b128), shared by all 4 waves.
// ---------------------------------------------------------------------------
__global__ __launch_bounds__(128) void conv_mix(const float* __restrict__ c,
                                                const float* __restrict__ convb,
                                                const unsigned short* __restrict__ wfrag,
                                                const float* __restrict__ hbuf,
                                                float* __restrict__ out) {
  int xt = blockIdx.x, y = blockIdx.y, b = blockIdx.z;
  __shared__ __align__(32) unsigned short BT[16 * KCONV];   // [pix][k] bf16, 18KB
  __shared__ float hstg[COUT * MD * 2];                     // h[o][kx] for (b,y)
  __shared__ float cosT[256], sinT[256];
  __shared__ float biasS[COUT];
  int t = threadIdx.x;
  for (int f = t; f < 256; f += 128) {
    float s, co; __sincosf(6.283185307179586f * f / 256.f, &s, &co);
    cosT[f] = co; sinT[f] = s;
  }
  if (t < COUT) biasS[t] = convb[t];
  const float* hp = hbuf + ((long)(b * H_ + y)) * COUT * MD * 2;
  for (int f = t; f < COUT * MD * 2; f += 128) hstg[f] = hp[f];

  int x0 = xt * 16;
  const float* cb = c + ((long)b) * CIN * H_ * W_;
  // Build BT: 16 pix x 36 segments of 16 consecutive k, vector-stored.
  for (int s = t; s < 16 * 36; s += 128) {
    int pix = s / 36, kg = s % 36, k0 = kg * 16;
    int i = k0 / 9, r = k0 % 9, dy = r / 3, dx = r - 3 * dy;
    Frag fr;
    #pragma unroll
    for (int j = 0; j < 16; ++j) {
      int yy = (y + dy - 1) & 255;
      int xx = (x0 + pix + dx - 1) & 255;              // circular padding
      fr.u[j] = f2bf(cb[((long)i * H_ + yy) * W_ + xx]);
      if (++dx == 3) { dx = 0; if (++dy == 3) { dy = 0; ++i; } }
    }
    *(v16u*)(&BT[pix * KCONV + k0]) = fr.u;
  }
  __syncthreads();

  int lane = t & 31, otile = t >> 5;
  int pix = lane & 15, lh = lane >> 4;
  v8f acc = {};
  const unsigned short* wf = wfrag + ((long)otile * KSTEPS) * 32 * 16;
  for (int ks = 0; ks < KSTEPS; ++ks) {
    Frag a, bb;
    a.u  = *(const v16u*)(wf + (ks * 32 + lane) * 16);           // prepacked 32B
    bb.u = *(const v16u*)(&BT[pix * KCONV + ks * 32 + lh * 16]); // contiguous 32B
    acc = __builtin_amdgcn_wmma_f32_16x16x32_bf16(false, a.b, false, bb.b, (short)0, acc, false, false);
  }

  // epilogue: bias, irfft along x from 16 modes, mix, store (coalesced in x)
  int x = x0 + pix;
  const float inv = 1.0f / ((float)H_ * (float)W_);
  #pragma unroll
  for (int v = 0; v < 8; ++v) {
    int o = otile * 16 + lh * 8 + v;
    float g = 0.f;
    #pragma unroll
    for (int kx = 0; kx < MD; ++kx) {
      int a2 = (kx * x) & 255;
      float hr = hstg[(o * MD + kx) * 2], hi = hstg[(o * MD + kx) * 2 + 1];
      float term = hr * cosT[a2] - hi * sinT[a2];     // Re(h * e^{+i theta})
      g += (kx == 0) ? term : 2.f * term;             // Hermitian doubling
    }
    g *= inv;
    float l = acc[v] + biasS[o];
    out[(((long)(b * COUT + o)) * H_ + y) * W_ + x] = g * l;
  }
}

// ---------------------------------------------------------------------------
extern "C" void kernel_launch(void* const* d_in, const int* in_sizes, int n_in,
                              void* d_out, int out_size, void* d_ws, size_t ws_size,
                              hipStream_t stream) {
  const float* c   = (const float*)d_in[0];
  const float* w1r = (const float*)d_in[1];
  const float* w1i = (const float*)d_in[2];
  const float* w2r = (const float*)d_in[3];
  const float* w2i = (const float*)d_in[4];
  const float* cw  = (const float*)d_in[5];
  const float* cb  = (const float*)d_in[6];
  float* out = (float*)d_out;

  char* ws = (char*)d_ws;
  unsigned short* xpk   = (unsigned short*)(ws + 0);              //  2 MB
  unsigned short* wspk  = (unsigned short*)(ws + (2ull  << 20));  //  8 MB
  float*          gout  = (float*)         (ws + (10ull << 20));  //  4 MB
  unsigned short* wfrag = (unsigned short*)(ws + (14ull << 20));  // 72 KB
  float*          hbuf  = (float*)         (ws + (15ull << 20));  // 32 MB

  hipLaunchKernelGGL(pack_conv_w, dim3((4 * KSTEPS * 32 * 16 + 255) / 256), dim3(256), 0, stream,
                     cw, wfrag);
  hipLaunchKernelGGL(pack_spec_w, dim3((NMODE * 2 * CIN * COUT + 255) / 256), dim3(256), 0, stream,
                     w1r, w1i, w2r, w2i, wspk);
  hipLaunchKernelGGL(dft_fwd, dim3(CIN, B_), dim3(256), 0, stream, c, xpk);
  hipLaunchKernelGGL(spectral, dim3(NMODE), dim3(128), 0, stream, xpk, wspk, gout);
  hipLaunchKernelGGL(idft_y, dim3(COUT, B_), dim3(256), 0, stream, gout, hbuf);
  hipLaunchKernelGGL(conv_mix, dim3(W_ / 16, H_, B_), dim3(128), 0, stream,
                     c, cb, wfrag, hbuf, out);
}
